// HeteroGraphSAGE_33208687133111
// MI455X (gfx1250) — matrix operand order repaired and checked by reading, not compiled, not run on hardware
//
#include <hip/hip_runtime.h>
#include <math.h>

// ---------------------------------------------------------------------------
// CDNA5 / gfx1250 wave32 WMMA bf16 path. D = H = 256, gate dim 1024, K=12.
// Weights pre-swizzled into WMMA B-fragment order (32B contiguous per lane).
// ---------------------------------------------------------------------------

typedef __attribute__((ext_vector_type(16))) __bf16 v16bf;
typedef __attribute__((ext_vector_type(8)))  float  v8f;

__device__ __forceinline__ float gelu_exact(float x) {
    return 0.5f * x * (1.0f + erff(x * 0.70710678118654752f));
}

// A fragment: 16x32 bf16, contiguous k = 16*hi + e (two ds_load_b128 per lane).
__device__ __forceinline__ void load_a_frag(const __bf16* lds, int kc, int ld,
                                            v16bf& a, int lane) {
    const int row = lane & 15;
    const int hi  = lane >> 4;
    const __bf16* base = lds + row * ld + kc;
#pragma unroll
    for (int e = 0; e < 16; ++e) {
        const int k  = hi * 16 + e;
        a[e] = base[k];
    }
}

// B fragment from pre-swizzled weights: fragment (ctile,kk) = 32 lanes x 16 bf16.
__device__ __forceinline__ v16bf load_bsw(const __bf16* __restrict__ Wsw,
                                          int ctile, int kk, int nkk, int lane) {
    return *(const v16bf*)(Wsw + ((size_t)(ctile * nkk + kk) << 9) + lane * 16);
}

#define WMMA_BF16(A, B, C) \
    __builtin_amdgcn_wmma_f32_16x16x32_bf16(false, (A), false, (B), (short)0, (C), false, false)

// ---------------------------------------------------------------------------
// Weight pre-swizzle: W (f32) -> bf16 B-fragment layout.
// transposed=1: B[k][j] = W[j*Ktot + k]   (x @ W^T, W is [Ncols,Ktot])
// transposed=0: B[k][j] = W[k*Ncols + j]  (x @ W,  W is [Ktot,Ncols])
// ---------------------------------------------------------------------------
__global__ __launch_bounds__(256) void preswz_kernel(const float* __restrict__ W,
                                                     __bf16* __restrict__ out,
                                                     int Ktot, int Ncols, int transposed)
{
    const int nkk = Ktot >> 5;
    const size_t o = (size_t)blockIdx.x * 256 + threadIdx.x;
    const int e  = (int)(o & 15);
    const int l  = (int)((o >> 4) & 31);
    const size_t r = o >> 9;
    const int kk = (int)(r % nkk);
    const int ct = (int)(r / nkk);
    const int col = ct * 16 + (l & 15);
    const int k   = kk * 32 + (l >> 4) * 16 + e;
    const float v = transposed ? W[(size_t)col * Ktot + k]
                               : W[(size_t)k * Ncols + col];
    out[o] = (__bf16)v;
}

// ---------------------------------------------------------------------------
// LSTM neighbor aggregation: 16 dst rows per block, knb-step LSTM.
// gates[16,1024] = x@Wih^T + h@Whh^T + (bih+bhh), via WMMA.
// Wave w owns gate column tiles w*8 .. w*8+7.
// ---------------------------------------------------------------------------
__global__ __launch_bounds__(256) void lstm_agg_kernel(
    const float* __restrict__ src, const int* __restrict__ nbr,
    const __bf16* __restrict__ WihSW, const __bf16* __restrict__ WhhSW,
    const float* __restrict__ bih, const float* __restrict__ bhh,
    float* __restrict__ hout, int Ndst, int knb)
{
    __shared__ __bf16 sAx[16 * 256];     // gathered x, bf16
    __shared__ __bf16 sAh[16 * 256];     // hidden state, bf16
    __shared__ float  sG[16 * 1024];     // gate pre-activations

    const int tid  = threadIdx.x;
    const int lane = tid & 31;
    const int wave = tid >> 5;
    const int n0   = blockIdx.x * 16;

    // per-lane combined bias for this wave's 8 column tiles (step-invariant)
    float biasreg[8];
#pragma unroll
    for (int ct = 0; ct < 8; ++ct) {
        const int col = (wave * 8 + ct) * 16 + (lane & 15);
        biasreg[ct] = bih[col] + bhh[col];
    }

    float creg[16];                       // cell state: thread tid owns column tid
#pragma unroll
    for (int j = 0; j < 16; ++j) { creg[j] = 0.0f; sAh[j * 256 + tid] = (__bf16)0.0f; }

#pragma unroll 1
    for (int step = 0; step < knb; ++step) {
        // Opaque zero offset defined inside the loop: blocks LICM from hoisting
        // the 128 B-fragment loads out of the step loop (which caused scratch
        // spills), while keeping global-pointer provenance (global_load+saddr).
        unsigned woff = 0;
        asm volatile("" : "+s"(woff));
        const __bf16* wih = WihSW + woff;
        const __bf16* whh = WhhSW + woff;

        // gather neighbor features (coalesced over tid = feature column)
#pragma unroll
        for (int j = 0; j < 16; ++j) {
            const int rr = n0 + j;
            const int g  = (rr < Ndst) ? nbr[(size_t)rr * knb + step] : 0;
            sAx[j * 256 + tid] = (__bf16)src[(size_t)g * 256 + tid];
        }
        __syncthreads();

        v8f acc[8];
#pragma unroll
        for (int ct = 0; ct < 8; ++ct)
#pragma unroll
            for (int r = 0; r < 8; ++r) acc[ct][r] = biasreg[ct];

        // x @ Wih^T  (one kk iteration in flight: 8 B frags + 1 A frag live)
#pragma unroll 1
        for (int kk = 0; kk < 8; ++kk) {
            v16bf a; load_a_frag(sAx, kk * 32, 256, a, lane);
            v16bf b[8];
#pragma unroll
            for (int ct = 0; ct < 8; ++ct) b[ct] = load_bsw(wih, wave * 8 + ct, kk, 8, lane);
#pragma unroll
            for (int ct = 0; ct < 8; ++ct) acc[ct] = WMMA_BF16(a, b[ct], acc[ct]);
        }
        // h @ Whh^T
#pragma unroll 1
        for (int kk = 0; kk < 8; ++kk) {
            v16bf a; load_a_frag(sAh, kk * 32, 256, a, lane);
            v16bf b[8];
#pragma unroll
            for (int ct = 0; ct < 8; ++ct) b[ct] = load_bsw(whh, wave * 8 + ct, kk, 8, lane);
#pragma unroll
            for (int ct = 0; ct < 8; ++ct) acc[ct] = WMMA_BF16(a, b[ct], acc[ct]);
        }

        // stage gates
        {
            const int hi = lane >> 4;
#pragma unroll
            for (int ct = 0; ct < 8; ++ct) {
                const int col = (wave * 8 + ct) * 16 + (lane & 15);
#pragma unroll
                for (int r = 0; r < 8; ++r) sG[(r + hi * 8) * 1024 + col] = acc[ct][r];
            }
        }
        __syncthreads();

        // elementwise LSTM cell update (torch gate order i,f,g,o)
        const bool last = (step == knb - 1);
#pragma unroll
        for (int j = 0; j < 16; ++j) {
            const float gi = sG[j * 1024 + tid];
            const float gf = sG[j * 1024 + 256 + tid];
            const float gg = sG[j * 1024 + 512 + tid];
            const float go = sG[j * 1024 + 768 + tid];
            const float si = 1.0f / (1.0f + expf(-gi));
            const float sf = 1.0f / (1.0f + expf(-gf));
            const float so = 1.0f / (1.0f + expf(-go));
            const float cc = sf * creg[j] + si * tanhf(gg);
            const float hh = so * tanhf(cc);
            creg[j] = cc;
            sAh[j * 256 + tid] = (__bf16)hh;
            if (last) {
                const int rr = n0 + j;
                if (rr < Ndst) hout[(size_t)rr * 256 + tid] = hh;
            }
        }
        __syncthreads();
    }
}

// ---------------------------------------------------------------------------
// out[n,:] (+)= gelu(dst[n,:]@Wself + hN[n,:]@Wneigh + bias)
// ---------------------------------------------------------------------------
template <bool ACC>
__global__ __launch_bounds__(256) void sage_combine_kernel(
    const float* __restrict__ dst, const float* __restrict__ hN,
    const __bf16* __restrict__ WselfSW, const __bf16* __restrict__ WneighSW,
    const float* __restrict__ bias, float* __restrict__ out, int N)
{
    __shared__ __bf16 sD[16 * 256];
    __shared__ __bf16 sN[16 * 256];

    const int tid  = threadIdx.x;
    const int lane = tid & 31;
    const int wave = tid >> 5;
    const int n0   = blockIdx.x * 16;

    for (int i = tid; i < 16 * 256; i += 256) {
        const int r = i >> 8, c = i & 255, rr = n0 + r;
        sD[i] = (__bf16)((rr < N) ? dst[(size_t)rr * 256 + c] : 0.0f);
        sN[i] = (__bf16)((rr < N) ? hN[(size_t)rr * 256 + c] : 0.0f);
    }
    __syncthreads();

    v8f acc[2];
#pragma unroll
    for (int ct = 0; ct < 2; ++ct) {
        const float bb = bias[(wave * 2 + ct) * 16 + (lane & 15)];
#pragma unroll
        for (int r = 0; r < 8; ++r) acc[ct][r] = bb;
    }

#pragma unroll 1
    for (int kk = 0; kk < 8; ++kk) {
        v16bf a; load_a_frag(sD, kk * 32, 256, a, lane);
        v16bf b[2];
#pragma unroll
        for (int ct = 0; ct < 2; ++ct) b[ct] = load_bsw(WselfSW, wave * 2 + ct, kk, 8, lane);
#pragma unroll
        for (int ct = 0; ct < 2; ++ct) acc[ct] = WMMA_BF16(a, b[ct], acc[ct]);
    }
#pragma unroll 1
    for (int kk = 0; kk < 8; ++kk) {
        v16bf a; load_a_frag(sN, kk * 32, 256, a, lane);
        v16bf b[2];
#pragma unroll
        for (int ct = 0; ct < 2; ++ct) b[ct] = load_bsw(WneighSW, wave * 2 + ct, kk, 8, lane);
#pragma unroll
        for (int ct = 0; ct < 2; ++ct) acc[ct] = WMMA_BF16(a, b[ct], acc[ct]);
    }

    const int hi = lane >> 4;
#pragma unroll
    for (int ct = 0; ct < 2; ++ct) {
        const int col = (wave * 2 + ct) * 16 + (lane & 15);
#pragma unroll
        for (int r = 0; r < 8; ++r) {
            const int rr = n0 + r + hi * 8;
            if (rr < N) {
                const float v = gelu_exact(acc[ct][r]);
                const size_t o = (size_t)rr * 256 + col;
                out[o] = ACC ? (out[o] + v) : v;
            }
        }
    }
}

// ---------------------------------------------------------------------------
// In-place LayerNorm over last dim (256), one block per row.
// ---------------------------------------------------------------------------
__global__ __launch_bounds__(256) void ln_kernel(float* __restrict__ x,
                                                 const float* __restrict__ g,
                                                 const float* __restrict__ b, int N)
{
    __shared__ float red[256];
    const int row = blockIdx.x;
    const int t   = threadIdx.x;
    if (row >= N) return;
    const float v = x[(size_t)row * 256 + t];
    red[t] = v; __syncthreads();
    for (int s = 128; s > 0; s >>= 1) { if (t < s) red[t] += red[t + s]; __syncthreads(); }
    const float mean = red[0] * (1.0f / 256.0f);
    __syncthreads();
    const float d = v - mean;
    red[t] = d * d; __syncthreads();
    for (int s = 128; s > 0; s >>= 1) { if (t < s) red[t] += red[t + s]; __syncthreads(); }
    const float var = red[0] * (1.0f / 256.0f);
    x[(size_t)row * 256 + t] = d * rsqrtf(var + 1e-5f) * g[t] + b[t];
}

// ---------------------------------------------------------------------------
// Deterministic two-pass BatchNorm statistics.
// ---------------------------------------------------------------------------
#define BN_ROWS 120
__global__ __launch_bounds__(256) void bn_partial_kernel(const float* __restrict__ x,
                                                         float* __restrict__ part, int N)
{
    const int t  = threadIdx.x;
    const int r0 = blockIdx.x * BN_ROWS;
    float s = 0.0f, s2 = 0.0f;
    for (int r = 0; r < BN_ROWS; ++r) {
        const int rr = r0 + r;
        if (rr < N) { const float v = x[(size_t)rr * 256 + t]; s += v; s2 += v * v; }
    }
    part[(size_t)blockIdx.x * 512 + t]       = s;
    part[(size_t)blockIdx.x * 512 + 256 + t] = s2;
}

__global__ __launch_bounds__(256) void bn_finalize_kernel(const float* __restrict__ part,
                                                          float* __restrict__ stats,
                                                          int nblk, int N)
{
    const int t = threadIdx.x;
    float s = 0.0f, s2 = 0.0f;
    for (int b = 0; b < nblk; ++b) {
        s  += part[(size_t)b * 512 + t];
        s2 += part[(size_t)b * 512 + 256 + t];
    }
    const float mean = s / (float)N;
    const float var  = s2 / (float)N - mean * mean;
    stats[t]       = mean;
    stats[256 + t] = rsqrtf(var + 1e-5f);
}

// ---------------------------------------------------------------------------
// Classifier: BN-apply -> Linear(256,256) via WMMA -> GELU -> Linear(256,1)
// ---------------------------------------------------------------------------
__global__ __launch_bounds__(256) void mlp_kernel(
    const float* __restrict__ x, const float* __restrict__ stats,
    const float* __restrict__ bng, const float* __restrict__ bnb,
    const __bf16* __restrict__ W1SW, const float* __restrict__ b1,
    const float* __restrict__ W2, const float* __restrict__ b2,
    float* __restrict__ out, int N)
{
    __shared__ __bf16 sA[16 * 256];
    __shared__ float  sY[16 * 256];
    __shared__ float  sRed[256];

    const int tid  = threadIdx.x;
    const int lane = tid & 31;
    const int wave = tid >> 5;
    const int n0   = blockIdx.x * 16;

    for (int i = tid; i < 16 * 256; i += 256) {
        const int r = i >> 8, c = i & 255, rr = n0 + r;
        float v = (rr < N) ? x[(size_t)rr * 256 + c] : 0.0f;
        v = (v - stats[c]) * stats[256 + c] * bng[c] + bnb[c];
        sA[i] = (__bf16)v;
    }
    __syncthreads();

    v8f acc[2];
#pragma unroll
    for (int ct = 0; ct < 2; ++ct) {
        const float bb = b1[(wave * 2 + ct) * 16 + (lane & 15)];
#pragma unroll
        for (int r = 0; r < 8; ++r) acc[ct][r] = bb;
    }
#pragma unroll 1
    for (int kk = 0; kk < 8; ++kk) {
        v16bf a; load_a_frag(sA, kk * 32, 256, a, lane);
        v16bf b[2];
#pragma unroll
        for (int ct = 0; ct < 2; ++ct) b[ct] = load_bsw(W1SW, wave * 2 + ct, kk, 8, lane);
#pragma unroll
        for (int ct = 0; ct < 2; ++ct) acc[ct] = WMMA_BF16(a, b[ct], acc[ct]);
    }
    const int hi = lane >> 4;
#pragma unroll
    for (int ct = 0; ct < 2; ++ct) {
        const int col = (wave * 2 + ct) * 16 + (lane & 15);
#pragma unroll
        for (int r = 0; r < 8; ++r) sY[(r + hi * 8) * 256 + col] = gelu_exact(acc[ct][r]);
    }
    __syncthreads();

    // out[row] = sY[row,:] . W2 + b2   (16 threads per row)
    const int row = tid >> 4, sub = tid & 15;
    float p = 0.0f;
#pragma unroll
    for (int j = 0; j < 16; ++j) { const int c = sub * 16 + j; p += sY[row * 256 + c] * W2[c]; }
    sRed[tid] = p; __syncthreads();
    if (sub == 0) {
        float s = 0.0f;
#pragma unroll
        for (int j = 0; j < 16; ++j) s += sRed[row * 16 + j];
        const int rr = n0 + row;
        if (rr < N) out[rr] = s + b2[0];
    }
}

// ---------------------------------------------------------------------------
// Launcher. Input order (setup_inputs insertion order, params flattened in
// insertion order): h_user, h_claim, nbr_pc, nbr_cc, nbr_cu,
//   then per conv {Wih,Whh,bih,bhh,Wself,Wneigh,b} for c1_pc,c1_cc,c1_cu,
//   c2_pc,c2_cc, then ln_g, ln_b, bn_g, bn_b, W1, b1, W2, b2.
// ---------------------------------------------------------------------------
extern "C" void kernel_launch(void* const* d_in, const int* in_sizes, int n_in,
                              void* d_out, int out_size, void* d_ws, size_t ws_size,
                              hipStream_t stream)
{
    const float* h_user  = (const float*)d_in[0];
    const float* h_claim = (const float*)d_in[1];
    const int*   nbr_pc  = (const int*)d_in[2];
    const int*   nbr_cc  = (const int*)d_in[3];
    const int*   nbr_cu  = (const int*)d_in[4];

    const int Nu = in_sizes[0] / 256;
    const int Nc = in_sizes[1] / 256;
    const int K  = in_sizes[2] / Nc;

    auto cw = [&](int conv, int j) { return (const float*)d_in[5 + conv * 7 + j]; };
    // conv: 0=c1_pc 1=c1_cc 2=c1_cu 3=c2_pc 4=c2_cc
    // j: 0=Wih 1=Whh 2=bih 3=bhh 4=Wself 5=Wneigh 6=b
    const float* ln_g = (const float*)d_in[40];
    const float* ln_b = (const float*)d_in[41];
    const float* bn_g = (const float*)d_in[42];
    const float* bn_b = (const float*)d_in[43];
    const float* W1   = (const float*)d_in[44];
    const float* b1   = (const float*)d_in[45];
    const float* W2   = (const float*)d_in[46];
    const float* b2   = (const float*)d_in[47];

    const int Nmax = (Nu > Nc) ? Nu : Nc;
    const size_t per = (size_t)Nmax * 256;
    float* agg   = (float*)d_ws;              // [Nmax,256] lstm aggregation scratch
    float* c1    = agg + per;                 // [Nc,256]
    float* u1    = c1 + per;                  // [Nu,256]
    float* c2    = u1 + per;                  // [Nc,256]
    float* part  = c2 + per;                  // BN partials
    const int nbn = (Nc + BN_ROWS - 1) / BN_ROWS;
    float* stats = part + (size_t)nbn * 512;  // [512]: mean, rstd

    // pre-swizzled bf16 weight region
    __bf16* swz = (__bf16*)(stats + 512);
    const size_t SZ_G = 256 * 1024;           // Wih/Whh swizzled (K=256, N=1024)
    const size_t SZ_S = 256 * 256;            // Wself/Wneigh/W1 swizzled
    const size_t PERC = 2 * SZ_G + 2 * SZ_S;  // per conv
    auto wihSW   = [&](int c) { return swz + (size_t)c * PERC; };
    auto whhSW   = [&](int c) { return swz + (size_t)c * PERC + SZ_G; };
    auto wselfSW = [&](int c) { return swz + (size_t)c * PERC + 2 * SZ_G; };
    auto wneighSW= [&](int c) { return swz + (size_t)c * PERC + 2 * SZ_G + SZ_S; };
    __bf16* w1SW = swz + 5 * PERC;

    dim3 blk(256);
    const int gbG = (int)(SZ_G >> 8);         // 1024 blocks
    const int gbS = (int)(SZ_S >> 8);         // 256 blocks

    // ---- weight pre-swizzle (cheap; L2-resident afterwards) ----
    for (int c = 0; c < 5; ++c) {
        preswz_kernel<<<gbG, blk, 0, stream>>>(cw(c,0), wihSW(c),   256, 1024, 1);
        preswz_kernel<<<gbG, blk, 0, stream>>>(cw(c,1), whhSW(c),   256, 1024, 1);
        preswz_kernel<<<gbS, blk, 0, stream>>>(cw(c,4), wselfSW(c), 256, 256,  0);
        preswz_kernel<<<gbS, blk, 0, stream>>>(cw(c,5), wneighSW(c),256, 256,  0);
    }
    preswz_kernel<<<gbS, blk, 0, stream>>>(W1, w1SW, 256, 256, 0);

    const int gbc = (Nc + 15) / 16;
    const int gbu = (Nu + 15) / 16;

    // ---- conv1 ----
    lstm_agg_kernel<<<gbc, blk, 0, stream>>>(h_user, nbr_pc, wihSW(0), whhSW(0), cw(0,2), cw(0,3), agg, Nc, K);
    sage_combine_kernel<false><<<gbc, blk, 0, stream>>>(h_claim, agg, wselfSW(0), wneighSW(0), cw(0,6), c1, Nc);
    lstm_agg_kernel<<<gbc, blk, 0, stream>>>(h_claim, nbr_cc, wihSW(1), whhSW(1), cw(1,2), cw(1,3), agg, Nc, K);
    sage_combine_kernel<true><<<gbc, blk, 0, stream>>>(h_claim, agg, wselfSW(1), wneighSW(1), cw(1,6), c1, Nc);
    lstm_agg_kernel<<<gbu, blk, 0, stream>>>(h_claim, nbr_cu, wihSW(2), whhSW(2), cw(2,2), cw(2,3), agg, Nu, K);
    sage_combine_kernel<false><<<gbu, blk, 0, stream>>>(h_user, agg, wselfSW(2), wneighSW(2), cw(2,6), u1, Nu);

    ln_kernel<<<Nc, blk, 0, stream>>>(c1, ln_g, ln_b, Nc);
    ln_kernel<<<Nu, blk, 0, stream>>>(u1, ln_g, ln_b, Nu);

    // ---- conv2 (claim output only) ----
    lstm_agg_kernel<<<gbc, blk, 0, stream>>>(u1, nbr_pc, wihSW(3), whhSW(3), cw(3,2), cw(3,3), agg, Nc, K);
    sage_combine_kernel<false><<<gbc, blk, 0, stream>>>(c1, agg, wselfSW(3), wneighSW(3), cw(3,6), c2, Nc);
    lstm_agg_kernel<<<gbc, blk, 0, stream>>>(c1, nbr_cc, wihSW(4), whhSW(4), cw(4,2), cw(4,3), agg, Nc, K);
    sage_combine_kernel<true><<<gbc, blk, 0, stream>>>(c1, agg, wselfSW(4), wneighSW(4), cw(4,6), c2, Nc);

    ln_kernel<<<Nc, blk, 0, stream>>>(c2, ln_g, ln_b, Nc);

    // ---- classifier ----
    bn_partial_kernel<<<nbn, blk, 0, stream>>>(c2, part, Nc);
    bn_finalize_kernel<<<1, blk, 0, stream>>>(part, stats, nbn, Nc);
    mlp_kernel<<<gbc, blk, 0, stream>>>(c2, stats, bn_g, bn_b, w1SW, b1, W2, b2, (float*)d_out, Nc);

    (void)n_in; (void)out_size; (void)ws_size;
}
